// DecoupleTaskInteraction_53343493816803
// MI455X (gfx1250) — matrix operand order repaired and checked by reading, not compiled, not run on hardware
//
#include <hip/hip_runtime.h>
#include <hip/hip_bf16.h>
#include <math.h>

typedef __bf16 bf16;
typedef __attribute__((ext_vector_type(16))) __bf16 v16bf;
typedef __attribute__((ext_vector_type(8)))  float   v8f;

#define LN_EPS 1e-5f
#define CD     256
#define HWD    4096
#define NBATCH 2

#define KLD  264   // padded K-chunk LDS row (bf16): 528B, bank step 4/lane
#define VTLD 40    // padded Vt-chunk LDS row (bf16): 80B, bank step 20/lane

static __device__ __forceinline__ v8f v8zero() {
  v8f a;
#pragma unroll
  for (int i = 0; i < 8; ++i) a[i] = 0.0f;
  return a;
}

static __device__ __forceinline__ v8f wmma_bf16(v16bf a, v16bf b, v8f c) {
  // (neg_a, A, neg_b, B, c_mod, C, reuse_a, reuse_b)
  return __builtin_amdgcn_wmma_f32_16x16x32_bf16(false, a, false, b, (short)0, c,
                                                 false, false);
}

// ---- CDNA5 ISA 7.12.2 tile loaders (wave32) --------------------------------
// A-matrix 16x32 bf16: lane l16 holds row M=l16; element e -> K = (e<8?e:e+8)+8*hlf
static __device__ __forceinline__ v16bf loadA_bf16(const bf16* S, int ld, int k0,
                                                   int hlf, int l16) {
  const bf16* p = S + (size_t)l16 * ld + k0 + hlf * 8;
  v16bf a;
#pragma unroll
  for (int e = 0; e < 16; ++e) a[e] = p[e < 8 ? e : e + 8];
  return a;
}

// B = M^T tile from bf16 row-major M: B[k][n] = M[n0+n][k0+k]
// lane reads 2 x 8 contiguous bf16 (K = (e<8?e:e+8)+8*hlf) -> 2 x b128
static __device__ __forceinline__ v16bf loadBT_bf16(const bf16* M, int ld, int n0,
                                                    int k0, int hlf, int l16) {
  const bf16* p = M + (size_t)(n0 + l16) * ld + k0 + hlf * 8;
  v16bf b;
#pragma unroll
  for (int e = 0; e < 16; ++e) b[e] = p[e < 8 ? e : e + 8];
  return b;
}

// reductions across the 16 lanes of a half-wave (masks < 16 stay in-half)
static __device__ __forceinline__ float redsum16(float v) {
#pragma unroll
  for (int m = 1; m < 16; m <<= 1) v += __shfl_xor(v, m, 32);
  return v;
}
static __device__ __forceinline__ float redmax16(float v) {
#pragma unroll
  for (int m = 1; m < 16; m <<= 1) v = fmaxf(v, __shfl_xor(v, m, 32));
  return v;
}

// async-stage one 32-key chunk of K (row-major) and Vt (chan-major) into LDS.
// 128 threads; each issues 16 GLOBAL_LOAD_ASYNC_TO_LDS_B128 (ASYNCcnt).
// INST_OFFSET is added to BOTH the LDS and global address (CDNA5 ISA 10.x).
static __device__ __forceinline__ void stage_chunk(const bf16* Kp, const bf16* VtB,
                                                   bf16* kb, bf16* vt, int j0,
                                                   int tid) {
  {  // K chunk: 32 rows x 512B; thread copies a 128B quarter of one row
    unsigned lds = (unsigned)(uintptr_t)(kb + (tid >> 2) * KLD + (tid & 3) * 64);
    const bf16* g = Kp + (size_t)(j0 + (tid >> 2)) * CD + (tid & 3) * 64;
#pragma unroll
    for (int i = 0; i < 8; ++i)
      asm volatile("global_load_async_to_lds_b128 %0, %1, off offset:%2"
                   :
                   : "v"(lds), "v"(g), "i"(i * 16)
                   : "memory");
  }
#pragma unroll
  for (int t = 0; t < 2; ++t) {  // Vt chunk: 256 rows x 64B; thread copies 2 rows
    int chan = tid * 2 + t;
    unsigned lds = (unsigned)(uintptr_t)(vt + chan * VTLD);
    const bf16* g = VtB + (size_t)chan * HWD + j0;
#pragma unroll
    for (int i = 0; i < 4; ++i)
      asm volatile("global_load_async_to_lds_b128 %0, %1, off offset:%2"
                   :
                   : "v"(lds), "v"(g), "i"(i * 16)
                   : "memory");
  }
}

// ============================================================================
// Kernel 0: one-shot f32 -> bf16 weight conversion
// ============================================================================
__global__ __launch_bounds__(256)
void dti_cvt_kernel(const float* __restrict__ src, bf16* __restrict__ dst, int n) {
  int i = blockIdx.x * 256 + threadIdx.x;
  if (i < n) dst[i] = (bf16)src[i];
}

// ============================================================================
// Kernel 1: per-token prep.  1 wave per 16-token tile.
//  - LN(center/wh/cls), mix = cat @ mix_w^T + b, LN(mix)
//  - K (row-major), Vt (TRANSPOSED: [b][chan][token]), Q1..Q3 in bf16
// ============================================================================
__global__ __launch_bounds__(32)
void dti_prep_kernel(const float* __restrict__ center, const float* __restrict__ whf,
                     const float* __restrict__ clsf,
                     const bf16* __restrict__ mixWb, const float* __restrict__ mix_b,
                     const float* __restrict__ ln_g, const float* __restrict__ ln_b,
                     const bf16* __restrict__ q1Wb, const float* __restrict__ q1_b,
                     const bf16* __restrict__ q2Wb, const float* __restrict__ q2_b,
                     const bf16* __restrict__ q3Wb, const float* __restrict__ q3_b,
                     const bf16* __restrict__ kWb, const float* __restrict__ k_b,
                     const bf16* __restrict__ vWb, const float* __restrict__ v_b,
                     bf16* __restrict__ Kb, bf16* __restrict__ Vt,
                     bf16* __restrict__ Q1, bf16* __restrict__ Q2,
                     bf16* __restrict__ Q3) {
  __shared__ bf16 catb[16 * 768];     // raw concatenated features (bf16)
  __shared__ bf16 nrm[3][16 * 256];   // LN'd streams
  __shared__ bf16 mtb[16 * 256];      // LN'd mix

  const int lane = threadIdx.x;
  const int hlf = lane >> 4, l16 = lane & 15;
  const int tile = blockIdx.x;        // 0..511
  const int bIdx = tile >> 8;
  const int p0 = (tile & 255) * 16;

  // ---- stage 1: load + LN the three input streams (2 lanes per row) ----
  const int mrow = lane >> 1;
  const int h2 = lane & 1;
  const float* srcs[3] = {center, whf, clsf};
#pragma unroll
  for (int s = 0; s < 3; ++s) {
    const float* x = srcs[s] + (size_t)bIdx * CD * HWD + (p0 + mrow);
    float sum = 0.f, sq = 0.f;
    for (int c = h2 * 128; c < h2 * 128 + 128; ++c) {
      float v = x[(size_t)c * HWD];
      sum += v;
      sq += v * v;
      catb[mrow * 768 + s * 256 + c] = (bf16)v;
    }
    sum += __shfl_xor(sum, 1, 32);
    sq += __shfl_xor(sq, 1, 32);
    float mean = sum * (1.f / 256.f);
    float rstd = rsqrtf(sq * (1.f / 256.f) - mean * mean + LN_EPS);
    for (int c = h2 * 128; c < h2 * 128 + 128; ++c) {
      float v = (float)catb[mrow * 768 + s * 256 + c];
      nrm[s][mrow * 256 + c] = (bf16)((v - mean) * rstd * ln_g[c] + ln_b[c]);
    }
  }
  __syncthreads();

  // ---- stage 2: mix GEMM [16x768]x[768x256] + LN(mix) ----
  v8f acc[16];
#pragma unroll
  for (int nt = 0; nt < 16; ++nt) {
    v8f a = v8zero();
#pragma unroll
    for (int k0 = 0; k0 < 768; k0 += 32)
      a = wmma_bf16(loadA_bf16(catb, 768, k0, hlf, l16),
                    loadBT_bf16(mixWb, 768, nt * 16, k0, hlf, l16), a);
    float bias = mix_b[nt * 16 + l16];
#pragma unroll
    for (int r = 0; r < 8; ++r) a[r] += bias;
    acc[nt] = a;
  }
#pragma unroll
  for (int r = 0; r < 8; ++r) {
    float sum = 0.f, sq = 0.f;
#pragma unroll
    for (int nt = 0; nt < 16; ++nt) {
      float v = acc[nt][r];
      sum += v;
      sq += v * v;
    }
    sum = redsum16(sum);
    sq = redsum16(sq);
    float mean = sum * (1.f / 256.f);
    float rstd = rsqrtf(sq * (1.f / 256.f) - mean * mean + LN_EPS);
    int M = r + 8 * hlf;
#pragma unroll
    for (int nt = 0; nt < 16; ++nt) {
      int n = nt * 16 + l16;
      mtb[M * 256 + n] = (bf16)((acc[nt][r] - mean) * rstd * ln_g[n] + ln_b[n]);
    }
  }
  __syncthreads();

  // ---- stage 3: K,V,Q1,Q2,Q3 projections ----
  const bf16* Asrc[5] = {mtb, mtb, nrm[0], nrm[1], nrm[2]};
  const bf16* Ws[5] = {kWb, vWb, q1Wb, q2Wb, q3Wb};
  const float* Bs[5] = {k_b, v_b, q1_b, q2_b, q3_b};
  bf16* Dst[5] = {Kb, (bf16*)0, Q1, Q2, Q3};
#pragma unroll
  for (int j = 0; j < 5; ++j) {
#pragma unroll
    for (int nt = 0; nt < 16; ++nt) {
      v8f a = v8zero();
#pragma unroll
      for (int k0 = 0; k0 < 256; k0 += 32)
        a = wmma_bf16(loadA_bf16(Asrc[j], 256, k0, hlf, l16),
                      loadBT_bf16(Ws[j], 256, nt * 16, k0, hlf, l16), a);
      float bias = Bs[j][nt * 16 + l16];
      int n = nt * 16 + l16;
      if (j == 1) {
        // V stored transposed: Vt[b][chan][token]; 8 consecutive tokens/lane
        bf16* vt = Vt + ((size_t)bIdx * CD + n) * HWD + p0 + 8 * hlf;
#pragma unroll
        for (int r = 0; r < 8; ++r) vt[r] = (bf16)(a[r] + bias);
      } else {
        bf16* dst = Dst[j] + ((size_t)bIdx * HWD + p0) * CD;
#pragma unroll
        for (int r = 0; r < 8; ++r)
          dst[(size_t)(r + 8 * hlf) * CD + n] = (bf16)(a[r] + bias);
      }
    }
  }
}

// ============================================================================
// Kernel 2: flash attention + residual + LN.
// 128 threads = 4 waves; each wave owns a 16-query tile (64 queries/block).
// K/Vt chunks (32 keys) are double-buffered in LDS via async loads shared by
// all 4 waves (4x less L2 traffic); next chunk's fetch overlaps current WMMAs.
// grid = (64 qblocks, 2 batch, 3 streams)
// ============================================================================
__global__ __launch_bounds__(128)
void dti_attn_kernel(const float* __restrict__ center, const float* __restrict__ whf,
                     const float* __restrict__ clsf,
                     const bf16* __restrict__ Kb, const bf16* __restrict__ Vt,
                     const bf16* __restrict__ Q1, const bf16* __restrict__ Q2,
                     const bf16* __restrict__ Q3,
                     const float* __restrict__ ln_g, const float* __restrict__ ln_b,
                     float* __restrict__ X0, float* __restrict__ X1,
                     float* __restrict__ X2) {
  __shared__ __align__(16) bf16 kbuf[2][32 * KLD];    // 2 x 16.5 KB
  __shared__ __align__(16) bf16 vtbuf[2][256 * VTLD]; // 2 x 20 KB
  __shared__ bf16 pb[4][16 * 32];                     // per-wave P bounce

  const int tid = threadIdx.x;
  const int wid = tid >> 5, lane = tid & 31;
  const int hlf = lane >> 4, l16 = lane & 15;
  const int q0 = blockIdx.x * 64 + wid * 16;
  const int bIdx = blockIdx.y;
  const int s = blockIdx.z;

  const bf16* Qs = (s == 0 ? Q1 : s == 1 ? Q2 : Q3) + ((size_t)bIdx * HWD + q0) * CD;
  const bf16* Kp = Kb + (size_t)bIdx * HWD * CD;
  const bf16* VtB = Vt + (size_t)bIdx * CD * HWD;

  // Q tile resident in registers (A-layout, 8 k-slices)
  v16bf aQ[8];
#pragma unroll
  for (int kt = 0; kt < 8; ++kt) aQ[kt] = loadA_bf16(Qs, 256, kt * 32, hlf, l16);

  v8f o[16];
#pragma unroll
  for (int nt = 0; nt < 16; ++nt) o[nt] = v8zero();
  float mrow[8], lrow[8];
#pragma unroll
  for (int r = 0; r < 8; ++r) {
    mrow[r] = -1e30f;
    lrow[r] = 0.f;
  }

  bf16* pw = pb[wid];
  stage_chunk(Kp, VtB, kbuf[0], vtbuf[0], 0, tid);   // prologue fetch

  for (int j0 = 0; j0 < HWD; j0 += 32) {
    const int cur = (j0 >> 5) & 1;
    if (j0 + 32 < HWD) {
      stage_chunk(Kp, VtB, kbuf[cur ^ 1], vtbuf[cur ^ 1], j0 + 32, tid);
      // async completes in order: <=16 outstanding retires the current chunk
      asm volatile("s_wait_asynccnt 0x10" ::: "memory");
    } else {
      asm volatile("s_wait_asynccnt 0x0" ::: "memory");
    }
    __syncthreads();  // all 4 waves' staged quarters visible

    const bf16* kb = kbuf[cur];
    const bf16* vt = vtbuf[cur];

    // S = Q K^T for 32 keys (two 16-col tiles); B tiles contiguous from LDS
    v8f s0 = v8zero(), s1 = v8zero();
#pragma unroll
    for (int kt = 0; kt < 8; ++kt) {
      s0 = wmma_bf16(aQ[kt], loadBT_bf16(kb, KLD, 0, kt * 32, hlf, l16), s0);
      s1 = wmma_bf16(aQ[kt], loadBT_bf16(kb, KLD, 16, kt * 32, hlf, l16), s1);
    }

    // online softmax (row M = r + 8*hlf, replicated across 16 lanes)
#pragma unroll
    for (int r = 0; r < 8; ++r) {
      float cmax = redmax16(fmaxf(s0[r], s1[r]));
      float mnew = fmaxf(mrow[r], cmax);
      float scale = __expf(mrow[r] - mnew);
      float p0v = __expf(s0[r] - mnew);
      float p1v = __expf(s1[r] - mnew);
      lrow[r] = lrow[r] * scale + redsum16(p0v + p1v);
      mrow[r] = mnew;
#pragma unroll
      for (int nt = 0; nt < 16; ++nt) o[nt][r] *= scale;
      int M = r + 8 * hlf;
      pw[M * 32 + l16] = (bf16)p0v;
      pw[M * 32 + 16 + l16] = (bf16)p1v;
    }

    // O += P V; Vt rows give 2 x 16B runs per lane (2 x ds_load_b128)
    v16bf aP = loadA_bf16(pw, 32, 0, hlf, l16);
#pragma unroll
    for (int nt = 0; nt < 16; ++nt)
      o[nt] = wmma_bf16(aP, loadBT_bf16(vt, VTLD, nt * 16, 0, hlf, l16), o[nt]);

    __syncthreads();  // everyone done reading cur before it is re-staged
  }

  // epilogue: normalize by row sum, residual, LayerNorm, store f32 to workspace
  const float* resid = (s == 0 ? center : s == 1 ? whf : clsf) + (size_t)bIdx * CD * HWD;
  float* Xo = (s == 0 ? X0 : s == 1 ? X1 : X2);
#pragma unroll
  for (int r = 0; r < 8; ++r) {
    float inv = 1.f / lrow[r];
    int M = r + 8 * hlf, p = q0 + M;
    float sum = 0.f, sq = 0.f;
#pragma unroll
    for (int nt = 0; nt < 16; ++nt) {
      int c = nt * 16 + l16;
      float v = o[nt][r] * inv + resid[(size_t)c * HWD + p];
      o[nt][r] = v;
      sum += v;
      sq += v * v;
    }
    sum = redsum16(sum);
    sq = redsum16(sq);
    float mean = sum * (1.f / 256.f);
    float rstd = rsqrtf(sq * (1.f / 256.f) - mean * mean + LN_EPS);
    float* row = Xo + ((size_t)bIdx * HWD + p) * CD;
#pragma unroll
    for (int nt = 0; nt < 16; ++nt) {
      int c = nt * 16 + l16;
      row[c] = (o[nt][r] - mean) * rstd * ln_g[c] + ln_b[c];
    }
  }
}

// ============================================================================
// Kernel 3: MLP (Linear -> exact GELU -> Linear) + residual + LN + untokenize.
// grid = (256 token tiles, 2 batch, 3 streams)
// ============================================================================
__global__ __launch_bounds__(32)
void dti_mlp_kernel(const float* __restrict__ X0, const float* __restrict__ X1,
                    const float* __restrict__ X2,
                    const bf16* __restrict__ cw1, const float* __restrict__ cb1,
                    const bf16* __restrict__ cw2, const float* __restrict__ cb2,
                    const bf16* __restrict__ ww1, const float* __restrict__ wb1,
                    const bf16* __restrict__ ww2, const float* __restrict__ wb2,
                    const bf16* __restrict__ lw1, const float* __restrict__ lb1,
                    const bf16* __restrict__ lw2, const float* __restrict__ lb2,
                    const float* __restrict__ ln_g, const float* __restrict__ ln_b,
                    float* __restrict__ out) {
  __shared__ bf16 xb[16 * 256];
  __shared__ bf16 hb[16 * 256];
  const int lane = threadIdx.x, hlf = lane >> 4, l16 = lane & 15;
  const int p0 = blockIdx.x * 16, bIdx = blockIdx.y, s = blockIdx.z;

  const float* X = (s == 0 ? X0 : s == 1 ? X1 : X2) + ((size_t)bIdx * HWD + p0) * CD;
  const bf16* W1 = (s == 0 ? cw1 : s == 1 ? ww1 : lw1);
  const float* B1 = (s == 0 ? cb1 : s == 1 ? wb1 : lb1);
  const bf16* W2 = (s == 0 ? cw2 : s == 1 ? ww2 : lw2);
  const float* B2 = (s == 0 ? cb2 : s == 1 ? wb2 : lb2);

  {
    int m = lane >> 1, h2 = lane & 1;
    const float* xr = X + (size_t)m * CD + h2 * 128;
    bf16* xw = xb + m * 256 + h2 * 128;
    for (int c = 0; c < 128; ++c) xw[c] = (bf16)xr[c];
  }
  __syncthreads();

  // GEMM1 + exact GELU -> hb
#pragma unroll
  for (int nt = 0; nt < 16; ++nt) {
    v8f a = v8zero();
#pragma unroll
    for (int k0 = 0; k0 < 256; k0 += 32)
      a = wmma_bf16(loadA_bf16(xb, 256, k0, hlf, l16),
                    loadBT_bf16(W1, 256, nt * 16, k0, hlf, l16), a);
    float bias = B1[nt * 16 + l16];
    int n = nt * 16 + l16;
#pragma unroll
    for (int r = 0; r < 8; ++r) {
      float v = a[r] + bias;
      v = 0.5f * v * (1.f + erff(v * 0.70710678118654752f));   // exact GELU
      hb[(r + 8 * hlf) * 256 + n] = (bf16)v;
    }
  }
  __syncthreads();

  // GEMM2
  v8f acc[16];
#pragma unroll
  for (int nt = 0; nt < 16; ++nt) {
    v8f a = v8zero();
#pragma unroll
    for (int k0 = 0; k0 < 256; k0 += 32)
      a = wmma_bf16(loadA_bf16(hb, 256, k0, hlf, l16),
                    loadBT_bf16(W2, 256, nt * 16, k0, hlf, l16), a);
    float bias = B2[nt * 16 + l16];
#pragma unroll
    for (int r = 0; r < 8; ++r) a[r] += bias;
    acc[nt] = a;
  }

  // residual + LN + transposed scatter to [stream][b][c][h*w]
  float* Os = out + (size_t)s * NBATCH * CD * HWD + (size_t)bIdx * CD * HWD;
#pragma unroll
  for (int r = 0; r < 8; ++r) {
    int M = r + 8 * hlf, p = p0 + M;
    float sum = 0.f, sq = 0.f;
#pragma unroll
    for (int nt = 0; nt < 16; ++nt) {
      int c = nt * 16 + l16;
      float v = acc[nt][r] + X[(size_t)M * CD + c];
      acc[nt][r] = v;
      sum += v;
      sq += v * v;
    }
    sum = redsum16(sum);
    sq = redsum16(sq);
    float mean = sum * (1.f / 256.f);
    float rstd = rsqrtf(sq * (1.f / 256.f) - mean * mean + LN_EPS);
#pragma unroll
    for (int nt = 0; nt < 16; ++nt) {
      int c = nt * 16 + l16;
      Os[(size_t)c * HWD + p] = (acc[nt][r] - mean) * rstd * ln_g[c] + ln_b[c];
    }
  }
}

// ============================================================================
extern "C" void kernel_launch(void* const* d_in, const int* in_sizes, int n_in,
                              void* d_out, int out_size, void* d_ws, size_t ws_size,
                              hipStream_t stream) {
  const float* center = (const float*)d_in[0];
  const float* whf    = (const float*)d_in[1];
  const float* clsf   = (const float*)d_in[2];
  const float* mix_w  = (const float*)d_in[3];
  const float* mix_b  = (const float*)d_in[4];
  const float* ln_g   = (const float*)d_in[5];
  const float* ln_b   = (const float*)d_in[6];
  const float* q1_w = (const float*)d_in[7],  *q1_b = (const float*)d_in[8];
  const float* q2_w = (const float*)d_in[9],  *q2_b = (const float*)d_in[10];
  const float* q3_w = (const float*)d_in[11], *q3_b = (const float*)d_in[12];
  const float* k_w  = (const float*)d_in[13], *k_b  = (const float*)d_in[14];
  const float* v_w  = (const float*)d_in[15], *v_b  = (const float*)d_in[16];
  const float* cw1 = (const float*)d_in[17], *cb1 = (const float*)d_in[18];
  const float* cw2 = (const float*)d_in[19], *cb2 = (const float*)d_in[20];
  const float* ww1 = (const float*)d_in[21], *wb1 = (const float*)d_in[22];
  const float* ww2 = (const float*)d_in[23], *wb2 = (const float*)d_in[24];
  const float* lw1 = (const float*)d_in[25], *lb1 = (const float*)d_in[26];
  const float* lw2 = (const float*)d_in[27], *lb2 = (const float*)d_in[28];
  float* out = (float*)d_out;

  // workspace: K, Vt, Q1..Q3 (bf16) + X0..X2 (f32) + bf16 weight copies
  char* ws = (char*)d_ws;
  const size_t szbf = (size_t)NBATCH * HWD * CD * sizeof(bf16);   // 4 MB
  const size_t szf  = (size_t)NBATCH * HWD * CD * sizeof(float);  // 8 MB
  bf16* Kb = (bf16*)ws;             ws += szbf;
  bf16* Vt = (bf16*)ws;             ws += szbf;
  bf16* Q1 = (bf16*)ws;             ws += szbf;
  bf16* Q2 = (bf16*)ws;             ws += szbf;
  bf16* Q3 = (bf16*)ws;             ws += szbf;
  float* X0 = (float*)ws;           ws += szf;
  float* X1 = (float*)ws;           ws += szf;
  float* X2 = (float*)ws;           ws += szf;

  const int nMix = 768 * CD, nSq = CD * CD;
  const float* wsrc[12] = {mix_w, k_w, v_w, q1_w, q2_w, q3_w,
                           cw1, cw2, ww1, ww2, lw1, lw2};
  bf16* wdst[12];
  int wn[12];
  for (int i = 0; i < 12; ++i) {
    wn[i] = (i == 0) ? nMix : nSq;
    wdst[i] = (bf16*)ws;
    ws += (size_t)wn[i] * sizeof(bf16);
  }

  for (int i = 0; i < 12; ++i)
    dti_cvt_kernel<<<dim3((wn[i] + 255) / 256), dim3(256), 0, stream>>>(
        wsrc[i], wdst[i], wn[i]);

  dti_prep_kernel<<<dim3(512), dim3(32), 0, stream>>>(
      center, whf, clsf, wdst[0], mix_b, ln_g, ln_b, wdst[3], q1_b, wdst[4], q2_b,
      wdst[5], q3_b, wdst[1], k_b, wdst[2], v_b, Kb, Vt, Q1, Q2, Q3);

  dti_attn_kernel<<<dim3(64, NBATCH, 3), dim3(128), 0, stream>>>(
      center, whf, clsf, Kb, Vt, Q1, Q2, Q3, ln_g, ln_b, X0, X1, X2);

  dti_mlp_kernel<<<dim3(256, NBATCH, 3), dim3(32), 0, stream>>>(
      X0, X1, X2, wdst[6], cb1, wdst[7], cb2, wdst[8], wb1, wdst[9], wb2,
      wdst[10], lb1, wdst[11], lb2, ln_g, ln_b, out);
}
